// CompositionalMHA_34316788695876
// MI455X (gfx1250) — matrix operand order repaired and checked by reading, not compiled, not run on hardware
//
#include <hip/hip_runtime.h>
#include <cstdint>

// ---------- types ----------
typedef __attribute__((ext_vector_type(16))) __bf16 bf16x16;
typedef __attribute__((ext_vector_type(8)))  float  f32x8;
typedef __attribute__((ext_vector_type(4)))  unsigned int u32x4;
typedef __attribute__((ext_vector_type(4)))  int v4i;

union FragB16 { u32x4 u4[2]; bf16x16 v; };

// ---------- async-LDS probe ----------
#if defined(__has_builtin)
#if __has_builtin(__builtin_amdgcn_global_load_async_to_lds_b128)
#define USE_ASYNC_LDS 1
#endif
#endif
#ifndef USE_ASYNC_LDS
#define USE_ASYNC_LDS 0
#endif

#if USE_ASYNC_LDS
// param0 = v4i32 addrspace(1)* (global src), param1 = v4i32 addrspace(3)* (LDS dst)
typedef __attribute__((address_space(1))) v4i* gv4i_p;
typedef __attribute__((address_space(3))) v4i* lv4i_p;
#define ASYNC_CP16(gsrc, ldst) \
  __builtin_amdgcn_global_load_async_to_lds_b128((gv4i_p)(gsrc), (lv4i_p)(ldst), 0, 0)
#define ASYNC_WAIT() asm volatile("s_wait_asynccnt 0" ::: "memory")
#else
#define ASYNC_CP16(gsrc, ldst) (*(u32x4*)(ldst) = *(const u32x4*)(gsrc))
#define ASYNC_WAIT()
#endif

// ---------- helpers ----------
__device__ __forceinline__ uint16_t f32_to_bf16(float f) {
  uint32_t u = __float_as_uint(f);
  uint32_t r = u + 0x7FFFu + ((u >> 16) & 1u);   // round-to-nearest-even
  return (uint16_t)(r >> 16);
}

__device__ __forceinline__ f32x8 zero8() {
  f32x8 z;
  for (int i = 0; i < 8; ++i) z[i] = 0.0f;
  return z;
}

__device__ __forceinline__ f32x8 wmma_bf16(bf16x16 a, bf16x16 b, f32x8 c) {
  return __builtin_amdgcn_wmma_f32_16x16x32_bf16(
      /*neg_a=*/false, a, /*neg_b=*/false, b,
      /*c_mod=*/(short)0, c, /*reuse_a=*/false, /*reuse_b=*/false);
}

// ---------- constants ----------
#define NT   4096      // B*S tokens
#define DM   1024      // d_model
#define KR   256       // top_k * r = 4*64
#define NH   16
#define HD   64
#define SEQ  1024

// =====================================================================
// 1) softmax + top-4 + renormalize (3 banks, 1 thread per bank)
// =====================================================================
__global__ void topk_kernel(const float* __restrict__ ql,
                            const float* __restrict__ kl,
                            const float* __restrict__ vl,
                            int* __restrict__ sel, float* __restrict__ wsel) {
  int b = threadIdx.x;
  if (b >= 3) return;
  const float* lg = (b == 0) ? ql : (b == 1) ? kl : vl;
  float w[16];
  float mx = -1e30f;
  for (int p = 0; p < 16; ++p) mx = fmaxf(mx, lg[p]);
  float s = 0.f;
  for (int p = 0; p < 16; ++p) { w[p] = __expf(lg[p] - mx); s += w[p]; }
  float inv = 1.0f / s;
  for (int p = 0; p < 16; ++p) w[p] *= inv;
  bool used[16];
  for (int p = 0; p < 16; ++p) used[p] = false;
  float tw[4]; float tsum = 0.f;
  for (int j = 0; j < 4; ++j) {
    int best = 0; float bv = -1e30f;
    for (int p = 0; p < 16; ++p)
      if (!used[p] && w[p] > bv) { bv = w[p]; best = p; }
    used[best] = true;
    sel[b * 4 + j] = best;
    tw[j] = bv; tsum += bv;
  }
  float rinv = 1.0f / tsum;
  for (int j = 0; j < 4; ++j) wsel[b * 4 + j] = tw[j] * rinv;
}

// =====================================================================
// 2) f32 -> bf16 convert / weight packing
// =====================================================================
__global__ void cvt_bf16_kernel(const float* __restrict__ in,
                                uint16_t* __restrict__ out, int n) {
  int i = blockIdx.x * blockDim.x + threadIdx.x;
  if (i < n) out[i] = f32_to_bf16(in[i]);
}

// Ut[j*64+rr][dd] = U[sel[j]][dd][rr]   (B-operand transpose of U_cat)
__global__ void pack_u_kernel(const float* __restrict__ U,
                              const int* __restrict__ sel,
                              uint16_t* __restrict__ Ut, int bank) {
  int tid = blockIdx.x * 256 + threadIdx.x;
  if (tid >= KR * DM) return;
  int dd = tid & (DM - 1);
  int rowk = tid >> 10;               // 0..255
  int j = rowk >> 6, rr = rowk & 63;
  int p = sel[bank * 4 + j];
  Ut[(size_t)rowk * DM + dd] =
      f32_to_bf16(U[((size_t)p * DM + dd) * 64 + rr]);
}

// Vt[dd][j*64+rr] = wsel[j] * V[sel[j]][rr][dd]  (B-operand transpose of V_cat)
__global__ void pack_v_kernel(const float* __restrict__ V,
                              const int* __restrict__ sel,
                              const float* __restrict__ wsel,
                              uint16_t* __restrict__ Vt, int bank) {
  int tid = blockIdx.x * 256 + threadIdx.x;
  if (tid >= DM * KR) return;
  int col = tid & (KR - 1);           // j*64+rr
  int dd = tid >> 8;
  int j = col >> 6, rr = col & 63;
  int p = sel[bank * 4 + j];
  float w = wsel[bank * 4 + j];
  Vt[(size_t)dd * KR + col] =
      f32_to_bf16(V[((size_t)p * 64 + rr) * DM + dd] * w);
}

// =====================================================================
// 3) bf16 WMMA GEMM: C[M,N] = A[M,K] * Bt[N,K]^T
//    256 thr (8 waves), 128x128 tile, K-step 32,
//    double-buffered LDS with async global->LDS copies.
// =====================================================================
#define TK 32
__global__ __launch_bounds__(256, 1) void gemm_bf16_kernel(
    const uint16_t* __restrict__ A, const uint16_t* __restrict__ Bt,
    void* __restrict__ Cp, int M, int N, int K,
    int lda, int ldb, int ldc, int out_bf16) {
  __shared__ uint16_t As[2][128 * TK];
  __shared__ uint16_t Bs[2][128 * TK];

  const int t = threadIdx.x;
  const int wave = t >> 5, lane = t & 31;
  const int g = lane >> 4, ln = lane & 15;
  const int m0 = blockIdx.y * 128, n0 = blockIdx.x * 128;
  const int wm = (wave >> 1) * 32, wn = (wave & 1) * 64;

  // fixed per-thread staging slots: rows (t>>2) and 64+(t>>2), col chunk (t&3)*8
  const int srow = t >> 2, scc = (t & 3) * 8;
  const uint16_t* gA0 = A + (size_t)(m0 + srow) * lda + scc;
  const uint16_t* gA1 = A + (size_t)(m0 + 64 + srow) * lda + scc;
  const uint16_t* gB0 = Bt + (size_t)(n0 + srow) * ldb + scc;
  const uint16_t* gB1 = Bt + (size_t)(n0 + 64 + srow) * ldb + scc;
  const int soff0 = srow * TK + scc;
  const int soff1 = (64 + srow) * TK + scc;

  f32x8 acc[2][4];
  for (int mi = 0; mi < 2; ++mi)
    for (int ni = 0; ni < 4; ++ni) acc[mi][ni] = zero8();

  // prologue: stage tile 0 into buffer 0
  ASYNC_CP16(gA0, &As[0][soff0]);
  ASYNC_CP16(gA1, &As[0][soff1]);
  ASYNC_CP16(gB0, &Bs[0][soff0]);
  ASYNC_CP16(gB1, &Bs[0][soff1]);
  ASYNC_WAIT();
  __syncthreads();

  int cur = 0;
  for (int k0 = 0; k0 < K; k0 += TK) {
    const bool more = (k0 + TK) < K;
    // issue next tile into the other buffer while computing this one
    if (more) {
      const int nk = k0 + TK;
      ASYNC_CP16(gA0 + nk, &As[cur ^ 1][soff0]);
      ASYNC_CP16(gA1 + nk, &As[cur ^ 1][soff1]);
      ASYNC_CP16(gB0 + nk, &Bs[cur ^ 1][soff0]);
      ASYNC_CP16(gB1 + nk, &Bs[cur ^ 1][soff1]);
    }

    FragB16 a[2], b[4];
    for (int mi = 0; mi < 2; ++mi) {
      int row = wm + mi * 16 + ln;
      a[mi].u4[0] = *(const u32x4*)&As[cur][row * TK + g * 8];
      a[mi].u4[1] = *(const u32x4*)&As[cur][row * TK + 16 + g * 8];
    }
    for (int ni = 0; ni < 4; ++ni) {
      int row = wn + ni * 16 + ln;
      b[ni].u4[0] = *(const u32x4*)&Bs[cur][row * TK + g * 8];
      b[ni].u4[1] = *(const u32x4*)&Bs[cur][row * TK + 16 + g * 8];
    }
    for (int mi = 0; mi < 2; ++mi)
      for (int ni = 0; ni < 4; ++ni)
        acc[mi][ni] = wmma_bf16(a[mi].v, b[ni].v, acc[mi][ni]);

    if (more) {
      ASYNC_WAIT();        // this wave's async copies landed
      __syncthreads();     // all waves done loading next + done reading cur
    }
    cur ^= 1;
  }

  // epilogue: C layout VGPR i -> row i+8g, col = lane&15
  // (uniform out_bf16 branch hoisted out of the loop nest)
  if (out_bf16) {
    uint16_t* C16 = (uint16_t*)Cp;
    for (int mi = 0; mi < 2; ++mi)
      for (int ni = 0; ni < 4; ++ni) {
        int col = n0 + wn + ni * 16 + ln;
        for (int i = 0; i < 8; ++i) {
          int r = m0 + wm + mi * 16 + i + 8 * g;
          C16[(size_t)r * ldc + col] = f32_to_bf16(acc[mi][ni][i]);
        }
      }
  } else {
    float* C32 = (float*)Cp;
    for (int mi = 0; mi < 2; ++mi)
      for (int ni = 0; ni < 4; ++ni) {
        int col = n0 + wn + ni * 16 + ln;
        for (int i = 0; i < 8; ++i) {
          int r = m0 + wm + mi * 16 + i + 8 * g;
          C32[(size_t)r * ldc + col] = acc[mi][ni][i];
        }
      }
  }
}

// =====================================================================
// 4) reorder q/k/v:  q,k -> [B,H,S,D] ; v -> [B,H,D,S] (pre-transposed B op)
// =====================================================================
__global__ void layout_qkv_kernel(const uint16_t* __restrict__ qb,
                                  const uint16_t* __restrict__ kb,
                                  const uint16_t* __restrict__ vb,
                                  uint16_t* __restrict__ Qh,
                                  uint16_t* __restrict__ Kh,
                                  uint16_t* __restrict__ Vta) {
  int tid = blockIdx.x * 256 + threadIdx.x;   // ((b*16+h)*1024+s)*64+d
  int d = tid & 63;
  int s = (tid >> 6) & 1023;
  int h = (tid >> 16) & 15;
  int b = tid >> 20;
  size_t src = ((size_t)(b * SEQ + s)) * DM + h * HD + d;
  Qh[tid] = qb[src];
  Kh[tid] = kb[src];
  Vta[((size_t)((b * NH + h) * HD + d)) * SEQ + s] = vb[src];
}

// =====================================================================
// 5) causal flash attention, one wave per 16 query rows
//    Q[B,H,S,64], K[B,H,S,64] (=Bt for QK^T), Vt[B,H,64,S] (=Bt for PV)
// =====================================================================
__global__ __launch_bounds__(128, 1) void attn_kernel(
    const uint16_t* __restrict__ Qh, const uint16_t* __restrict__ Kh,
    const uint16_t* __restrict__ Vta, uint16_t* __restrict__ O) {
  __shared__ uint16_t Ps[4][16 * 32];

  const int wib = threadIdx.x >> 5;
  const int wid = blockIdx.x * 4 + wib;
  const int lane = threadIdx.x & 31;
  const int g = lane >> 4, ln = lane & 15;
  const int qt = wid & 63;
  const int h = (wid >> 6) & 15;
  const int b = wid >> 10;
  const int qbase = qt * 16;

  const uint16_t* Qbh = Qh + ((size_t)(b * NH + h) * SEQ) * HD;
  const uint16_t* Kbh = Kh + ((size_t)(b * NH + h) * SEQ) * HD;
  const uint16_t* Vbh = Vta + ((size_t)(b * NH + h) * HD) * SEQ;
  uint16_t* myPs = Ps[wib];

  // Q fragments (K-dim = head_dim 64 -> 2 chunks of 32)
  FragB16 aq[2];
  {
    int row = qbase + ln;
    for (int c = 0; c < 2; ++c) {
      aq[c].u4[0] = *(const u32x4*)&Qbh[(size_t)row * HD + c * 32 + g * 8];
      aq[c].u4[1] = *(const u32x4*)&Qbh[(size_t)row * HD + c * 32 + 16 + g * 8];
    }
  }

  float mrow[8], lrow[8];
  for (int i = 0; i < 8; ++i) { mrow[i] = -1e30f; lrow[i] = 0.0f; }
  f32x8 acc[4];
  for (int nt = 0; nt < 4; ++nt) acc[nt] = zero8();
  const float scale = 0.125f;   // 1/sqrt(64)

  for (int kbk = 0; kbk < qbase + 16; kbk += 32) {
    // ---- scores: two 16x16 tiles over 32 keys ----
    f32x8 s[2];
    for (int sn = 0; sn < 2; ++sn) {
      int kbb = kbk + sn * 16;
      f32x8 sc = zero8();
      for (int c = 0; c < 2; ++c) {
        FragB16 bk;
        bk.u4[0] = *(const u32x4*)&Kbh[(size_t)(kbb + ln) * HD + c * 32 + g * 8];
        bk.u4[1] = *(const u32x4*)&Kbh[(size_t)(kbb + ln) * HD + c * 32 + 16 + g * 8];
        sc = wmma_bf16(aq[c].v, bk.v, sc);
      }
      s[sn] = sc;
    }
    // ---- mask + online softmax update ----
    for (int i = 0; i < 8; ++i) {
      int qi = qbase + i + 8 * g;
      float v0 = (kbk + ln <= qi) ? s[0][i] * scale : -1e30f;
      float v1 = (kbk + 16 + ln <= qi) ? s[1][i] * scale : -1e30f;
      float mx = fmaxf(v0, v1);
      for (int off = 1; off < 16; off <<= 1)
        mx = fmaxf(mx, __shfl_xor(mx, off, 32));
      float mnew = fmaxf(mrow[i], mx);
      float corr = __expf(mrow[i] - mnew);
      float p0 = __expf(v0 - mnew);
      float p1 = __expf(v1 - mnew);
      float rs = p0 + p1;
      for (int off = 1; off < 16; off <<= 1)
        rs += __shfl_xor(rs, off, 32);
      lrow[i] = lrow[i] * corr + rs;
      mrow[i] = mnew;
      for (int nt = 0; nt < 4; ++nt) acc[nt][i] *= corr;
      int r = i + 8 * g;
      myPs[r * 32 + ln] = f32_to_bf16(p0);
      myPs[r * 32 + 16 + ln] = f32_to_bf16(p1);
    }
    __builtin_amdgcn_wave_barrier();  // DS ops in-order within wave

    // ---- P (A-frag) from LDS, V (B-frag) from Vt, accumulate ----
    FragB16 pf;
    pf.u4[0] = *(const u32x4*)&myPs[ln * 32 + g * 8];
    pf.u4[1] = *(const u32x4*)&myPs[ln * 32 + 16 + g * 8];
    for (int nt = 0; nt < 4; ++nt) {
      FragB16 bv;
      bv.u4[0] = *(const u32x4*)&Vbh[(size_t)(nt * 16 + ln) * SEQ + kbk + g * 8];
      bv.u4[1] = *(const u32x4*)&Vbh[(size_t)(nt * 16 + ln) * SEQ + kbk + 16 + g * 8];
      acc[nt] = wmma_bf16(pf.v, bv.v, acc[nt]);
    }
  }

  // ---- write O back to [B*S, H*D] bf16 ----
  for (int nt = 0; nt < 4; ++nt)
    for (int i = 0; i < 8; ++i) {
      int r = qbase + i + 8 * g;
      float o = acc[nt][i] / lrow[i];
      O[((size_t)(b * SEQ + r)) * DM + h * HD + nt * 16 + ln] = f32_to_bf16(o);
    }
}

// =====================================================================
// launch
// =====================================================================
extern "C" void kernel_launch(void* const* d_in, const int* in_sizes, int n_in,
                              void* d_out, int out_size, void* d_ws, size_t ws_size,
                              hipStream_t stream) {
  const float* x  = (const float*)d_in[0];
  const float* Ub[3] = { (const float*)d_in[1], (const float*)d_in[3], (const float*)d_in[5] };
  const float* Vb[3] = { (const float*)d_in[2], (const float*)d_in[4], (const float*)d_in[6] };
  const float* ql = (const float*)d_in[7];
  const float* kl = (const float*)d_in[8];
  const float* vl = (const float*)d_in[9];
  const float* ow = (const float*)d_in[10];
  (void)in_sizes; (void)n_in; (void)out_size; (void)ws_size;

  char* ws = (char*)d_ws;
  size_t off = 0;
  auto alloc = [&](size_t bytes) -> void* {
    void* p = ws + off;
    off = (off + bytes + 255) & ~(size_t)255;
    return p;
  };

  int*      sel  = (int*)alloc(12 * sizeof(int));
  float*    wsel = (float*)alloc(12 * sizeof(float));
  uint16_t* xb   = (uint16_t*)alloc((size_t)NT * DM * 2);        // 8 MB
  uint16_t* Ut   = (uint16_t*)alloc((size_t)3 * KR * DM * 2);    // 1.5 MB
  uint16_t* Vt   = (uint16_t*)alloc((size_t)3 * DM * KR * 2);    // 1.5 MB
  uint16_t* owb  = (uint16_t*)alloc((size_t)DM * DM * 2);        // 2 MB
  uint16_t* Hb   = (uint16_t*)alloc((size_t)3 * NT * KR * 2);    // 6 MB
  uint16_t* qkv  = (uint16_t*)alloc((size_t)3 * NT * DM * 2);    // 24 MB
  uint16_t* Qh   = (uint16_t*)alloc((size_t)NT * DM * 2);        // 8 MB
  uint16_t* Kh   = (uint16_t*)alloc((size_t)NT * DM * 2);        // 8 MB
  uint16_t* Vta  = (uint16_t*)alloc((size_t)NT * DM * 2);        // 8 MB
  uint16_t* Oc   = (uint16_t*)alloc((size_t)NT * DM * 2);        // 8 MB

  // 1) top-k weights
  topk_kernel<<<1, 32, 0, stream>>>(ql, kl, vl, sel, wsel);

  // 2) conversions + packing
  cvt_bf16_kernel<<<(NT * DM + 255) / 256, 256, 0, stream>>>(x, xb, NT * DM);
  cvt_bf16_kernel<<<(DM * DM + 255) / 256, 256, 0, stream>>>(ow, owb, DM * DM);
  for (int bank = 0; bank < 3; ++bank) {
    pack_u_kernel<<<(KR * DM + 255) / 256, 256, 0, stream>>>(
        Ub[bank], sel, Ut + (size_t)bank * KR * DM, bank);
    pack_v_kernel<<<(DM * KR + 255) / 256, 256, 0, stream>>>(
        Vb[bank], sel, wsel, Vt + (size_t)bank * DM * KR, bank);
  }

  // 3) H_b = xb @ U_cat   (M=4096, N=256, K=1024)
  for (int bank = 0; bank < 3; ++bank) {
    gemm_bf16_kernel<<<dim3(KR / 128, NT / 128), 256, 0, stream>>>(
        xb, Ut + (size_t)bank * KR * DM, Hb + (size_t)bank * NT * KR,
        NT, KR, DM, DM, DM, KR, /*out_bf16=*/1);
  }
  // 4) qkv_b = H_b @ V_cat (M=4096, N=1024, K=256)
  for (int bank = 0; bank < 3; ++bank) {
    gemm_bf16_kernel<<<dim3(DM / 128, NT / 128), 256, 0, stream>>>(
        Hb + (size_t)bank * NT * KR, Vt + (size_t)bank * DM * KR,
        qkv + (size_t)bank * NT * DM,
        NT, DM, KR, KR, KR, DM, /*out_bf16=*/1);
  }

  // 5) head layout (q,k -> [B,H,S,D]; v -> [B,H,D,S])
  layout_qkv_kernel<<<(NT * DM) / 256, 256, 0, stream>>>(
      qkv, qkv + (size_t)NT * DM, qkv + (size_t)2 * NT * DM, Qh, Kh, Vta);

  // 6) causal attention (4096 waves, 4 per block)
  attn_kernel<<<(4 * NH * (SEQ / 16)) / 4, 128, 0, stream>>>(Qh, Kh, Vta, Oc);

  // 7) final projection: d_out = O @ out_w^T (f32 out)
  gemm_bf16_kernel<<<dim3(DM / 128, NT / 128), 256, 0, stream>>>(
      Oc, owb, d_out, NT, DM, DM, DM, DM, DM, /*out_bf16=*/0);
}
